// PMG_33517924778320
// MI455X (gfx1250) — compile-verified, matrix operand order checked by reading, not run on hardware
//
#include <hip/hip_runtime.h>
#include <hip/hip_bf16.h>
#include <math.h>

// ---------------- CDNA5 WMMA types ----------------
typedef __attribute__((ext_vector_type(16))) __bf16 v16bf;
typedef __attribute__((ext_vector_type(8)))  float  v8f;
typedef __attribute__((ext_vector_type(4)))  float  v4f;

// ---------------- problem constants ----------------
#define BATCH 4
#define TOPN  4
#define NPART 16            // BATCH*TOPN
#define CH    2048
#define CH2   1024
#define CH4   512
#define FEATD 512
#define CLS   200
#define IMS   448
#define PADW  224
#define LPADW 896
#define NANCH 1614
#define EPSBN 1e-5f
#define IOUT  0.25f

// ---------------- GEMM tile config ----------------
#define BM 64            // block M tile
#define BN 128           // block N tile
#define BK 32            // K step (one bf16 WMMA depth)
#define LDR 48           // LDS row stride in bf16 elems (96B, keeps 32B alignment)

// ============================================================
// Implicit-im2col GEMM with bf16 WMMA (f32 accumulate).
//   Y = act((W*im2col(X) + bias) * scale*smul + shift), input bn fold via iscale/ishift
//   8 waves, each computes a 32x32 C tile (4 accumulators, 4 WMMAs per K-step).
//   LDS tiles stored "fragment-ready": K pre-permuted so each lane's 16-elem
//   fragment is one contiguous 32B chunk -> ds_load_b128 pairs.
//   Global weight staging: 2x b128 loads/thread, double-buffered in registers.
// act: 0=none 1=relu 2=elu
// ============================================================
__global__ __launch_bounds__(256)
void gemm_conv_wmma(const float* __restrict__ Wt, const float* __restrict__ X,
                    float* __restrict__ Y,
                    const float* __restrict__ bias,
                    const float* __restrict__ scale, const float* __restrict__ shift,
                    float smul,
                    const float* __restrict__ iscale, const float* __restrict__ ishift,
                    float imul, int act,
                    int M, int Cin, int H, int Wd, int KH, int KW, int stride, int pad,
                    int Bn, int OH, int OW, int obs, int ooff)
{
    const int K   = Cin * KH * KW;
    const int HWo = OH * OW;
    const int N   = Bn * HWo;
    const int khw = KH * KW;

    __shared__ __attribute__((aligned(64))) __bf16 Af[BM * LDR];   // [m][kperm]
    __shared__ __attribute__((aligned(64))) __bf16 Bf[BN * LDR];   // [n][kperm]

    const int tid  = threadIdx.x;
    const int lane = tid & 31;
    const int wv   = tid >> 5;          // wave 0..7
    const int wm   = wv >> 2;           // 0..1  -> 32-row M sub-block
    const int wn   = wv & 3;            // 0..3  -> 32-col N sub-block
    const int mBase = blockIdx.y * BM;
    const int nBase = blockIdx.x * BN;
    const int r  = lane & 15;
    const int hi = lane >> 4;

    // ---- staging assignments ----
    // A: 64 rows x 32 k -> 4 threads/row, 8 k each (exactly one permute group)
    const int aRow = tid >> 2;
    const int aG   = tid & 3;
    const int aPos = (((aG & 1) << 1) | (aG >> 1)) * 8;      // g0->0, g1->2, g2->1, g3->3
    // B: 128 rows x 32 k -> 2 threads/row, 16 k (two permute groups)
    const int bRow = tid >> 1;
    const int bG0  = (tid & 1) * 2;                          // groups {0,1} or {2,3}
    // per-thread fixed n-decomposition for the im2col gather
    const int bN   = nBase + bRow;
    int nb = 0, noh = 0, now = 0;
    {
        int p = 0;
        if (bN < N) { nb = bN / HWo; p = bN - nb * HWo; noh = p / OW; now = p - noh * OW; }
    }

    float aReg[8];
    float bReg[16];

    auto loadA = [&](int k0) {
        int m = mBase + aRow;
        int kb = k0 + aG * 8;
        if (m < M && kb + 8 <= K) {
            const v4f* p = (const v4f*)(Wt + (size_t)m * K + kb);
            v4f u0 = p[0], u1 = p[1];
            aReg[0] = u0.x; aReg[1] = u0.y; aReg[2] = u0.z; aReg[3] = u0.w;
            aReg[4] = u1.x; aReg[5] = u1.y; aReg[6] = u1.z; aReg[7] = u1.w;
        } else {
            #pragma unroll
            for (int j = 0; j < 8; ++j) {
                int k = kb + j;
                aReg[j] = (m < M && k < K) ? Wt[(size_t)m * K + k] : 0.f;
            }
        }
    };

    auto loadB = [&](int k0) {
        #pragma unroll
        for (int gg = 0; gg < 2; ++gg) {
            #pragma unroll
            for (int j = 0; j < 8; ++j) {
                int k = k0 + (bG0 + gg) * 8 + j;
                float v = 0.f;
                if (k < K && bN < N) {
                    int ci = k / khw, rem = k - ci * khw;
                    int kh = rem / KW, kw = rem - kh * KW;
                    int ih = noh * stride - pad + kh;
                    int iw = now * stride - pad + kw;
                    if (ih >= 0 && ih < H && iw >= 0 && iw < Wd)
                        v = X[(((size_t)nb * Cin + ci) * H + ih) * Wd + iw];
                    if (iscale) v = v * iscale[ci] * imul;
                    if (ishift) v = v + ishift[ci];
                }
                bReg[gg * 8 + j] = v;
            }
        }
    };

    auto storeAB = [&]() {
        #pragma unroll
        for (int j = 0; j < 8; ++j)
            Af[aRow * LDR + aPos + j] = (__bf16)aReg[j];
        #pragma unroll
        for (int gg = 0; gg < 2; ++gg) {
            int g = bG0 + gg;
            int pos = (((g & 1) << 1) | (g >> 1)) * 8;
            #pragma unroll
            for (int j = 0; j < 8; ++j)
                Bf[bRow * LDR + pos + j] = (__bf16)bReg[gg * 8 + j];
        }
    };

    v8f acc00 = {0.f,0.f,0.f,0.f,0.f,0.f,0.f,0.f};
    v8f acc01 = acc00, acc10 = acc00, acc11 = acc00;

    loadA(0); loadB(0);
    storeAB();
    __syncthreads();

    const v16bf* Ap = (const v16bf*)Af;    // 16 bf16 per element, LDR/16 = 3 per row
    const v16bf* Bp = (const v16bf*)Bf;

    int k0 = 0;
    for (;;) {
        int k1 = k0 + BK;
        bool more = (k1 < K);
        if (more) { loadA(k1); loadB(k1); }   // prefetch next tile into registers

        v16bf a0 = Ap[(wm * 32 +      r) * 3 + hi];
        v16bf a1 = Ap[(wm * 32 + 16 + r) * 3 + hi];
        v16bf b0 = Bp[(wn * 32 +      r) * 3 + hi];
        v16bf b1 = Bp[(wn * 32 + 16 + r) * 3 + hi];
        acc00 = __builtin_amdgcn_wmma_f32_16x16x32_bf16(false, a0, false, b0, (short)0, acc00, false, false);
        acc01 = __builtin_amdgcn_wmma_f32_16x16x32_bf16(false, a0, false, b1, (short)0, acc01, false, false);
        acc10 = __builtin_amdgcn_wmma_f32_16x16x32_bf16(false, a1, false, b0, (short)0, acc10, false, false);
        acc11 = __builtin_amdgcn_wmma_f32_16x16x32_bf16(false, a1, false, b1, (short)0, acc11, false, false);

        if (!more) break;
        __syncthreads();     // all waves done reading LDS
        storeAB();           // commit prefetched tile
        __syncthreads();
        k0 = k1;
    }

    // ---- epilogue: C/D layout -> m_local = s*16 + 8*hi + j, n_local = t*16 + r ----
    #pragma unroll
    for (int s = 0; s < 2; ++s) {
        #pragma unroll
        for (int t = 0; t < 2; ++t) {
            v8f accv = (s == 0) ? (t == 0 ? acc00 : acc01)
                                : (t == 0 ? acc10 : acc11);
            int n = nBase + wn * 32 + t * 16 + r;
            if (n < N) {
                int b = n / HWo, p = n - b * HWo;
                #pragma unroll
                for (int j = 0; j < 8; ++j) {
                    int m = mBase + wm * 32 + s * 16 + hi * 8 + j;
                    if (m < M) {
                        float v = accv[j];
                        if (bias)  v += bias[m];
                        if (scale) v = v * (scale[m] * smul);
                        if (shift) v += shift[m];
                        if (act == 1)      v = fmaxf(v, 0.f);
                        else if (act == 2) v = (v > 0.f) ? v : expm1f(v);
                        Y[(size_t)b * obs + ooff + (size_t)m * HWo + p] = v;
                    }
                }
            }
        }
    }
}

// ============================================================
// Anchor generation (NTS-Net default anchors, padded + floored + clipped)
// ============================================================
__global__ void anchors_kernel(float* __restrict__ anch)
{
    int a = blockIdx.x * blockDim.x + threadIdx.x;
    if (a >= NANCH) return;
    const float sc2[2] = {1.2599210498948732f, 1.5874010519681994f};
    const float sc3[3] = {1.0f, 1.2599210498948732f, 1.5874010519681994f};
    const float ra[3]  = {2.f / 3.f, 1.f, 1.5f};
    int g, bi, pos; float stridef, sizef, s;
    if (a < 1176)      { g = 14; stridef = 32.f;  sizef = 48.f;  int a2 = a;        bi = a2 / 196; pos = a2 % 196; s = sc2[bi / 3]; }
    else if (a < 1470) { g = 7;  stridef = 64.f;  sizef = 96.f;  int a2 = a - 1176; bi = a2 / 49;  pos = a2 % 49;  s = sc2[bi / 3]; }
    else               { g = 4;  stridef = 112.f; sizef = 192.f; int a2 = a - 1470; bi = a2 / 16;  pos = a2 % 16;  s = sc3[bi / 3]; }
    float rr = ra[bi % 3];
    int iy = pos / g, ix = pos % g;
    float cy = (iy + 0.5f) * stridef, cx = (ix + 0.5f) * stridef;
    float h = sizef * s / sqrtf(rr), w = sizef * s * sqrtf(rr);
    float v[4] = {cy - h * 0.5f, cx - w * 0.5f, cy + h * 0.5f, cx + w * 0.5f};
    #pragma unroll
    for (int j = 0; j < 4; ++j)
        anch[a * 4 + j] = fminf(fmaxf(floorf(v[j] + (float)PADW), 0.f), (float)LPADW);
}

// ============================================================
// Hard NMS: one block per image, TOPN rounds of argmax + IoU suppression
// ============================================================
__global__ __launch_bounds__(256)
void nms_kernel(const float* __restrict__ rpn, const float* __restrict__ anch,
                int* __restrict__ idx_out, float* __restrict__ prob_out)
{
    int b = blockIdx.x;
    __shared__ float sc[NANCH];
    __shared__ float rv[256];
    __shared__ int   ri[256];
    int tid = threadIdx.x;
    for (int i = tid; i < NANCH; i += 256) sc[i] = rpn[b * NANCH + i];
    __syncthreads();
    for (int round = 0; round < TOPN; ++round) {
        float bv = -INFINITY; int bi = NANCH;
        for (int i = tid; i < NANCH; i += 256) {
            float v = sc[i];
            if (v > bv || (v == bv && i < bi)) { bv = v; bi = i; }
        }
        rv[tid] = bv; ri[tid] = bi;
        __syncthreads();
        for (int s = 128; s > 0; s >>= 1) {
            if (tid < s) {
                float ov = rv[tid + s]; int oi = ri[tid + s];
                if (ov > rv[tid] || (ov == rv[tid] && oi < ri[tid])) { rv[tid] = ov; ri[tid] = oi; }
            }
            __syncthreads();
        }
        int sel = ri[0]; float selv = rv[0];
        if (tid == 0) { idx_out[b * TOPN + round] = sel; prob_out[b * TOPN + round] = selv; }
        float by0 = anch[sel * 4 + 0], bx0 = anch[sel * 4 + 1];
        float by1 = anch[sel * 4 + 2], bx1 = anch[sel * 4 + 3];
        float barea = (by1 - by0) * (bx1 - bx0);
        for (int i = tid; i < NANCH; i += 256) {
            float y0 = anch[i * 4 + 0], x0 = anch[i * 4 + 1];
            float y1 = anch[i * 4 + 2], x1 = anch[i * 4 + 3];
            float area = (y1 - y0) * (x1 - x0);
            float yy0 = fmaxf(by0, y0), xx0 = fmaxf(bx0, x0);
            float yy1 = fminf(by1, y1), xx1 = fminf(bx1, x1);
            float inter = fmaxf(yy1 - yy0, 0.f) * fmaxf(xx1 - xx0, 0.f);
            float iou = inter / (area + barea - inter + 1e-9f);
            if (iou > IOUT) sc[i] = -INFINITY;
        }
        __syncthreads();
    }
}

// ============================================================
// Bilinear crop-resize 224x224 (align_corners), implicit zero pad
// ============================================================
__global__ void crop_kernel(const float* __restrict__ x, const float* __restrict__ anch,
                            const int* __restrict__ idx, float* __restrict__ parts)
{
    long gid = (long)blockIdx.x * blockDim.x + threadIdx.x;
    const long total = (long)NPART * 3 * 224 * 224;
    if (gid >= total) return;
    int ox = (int)(gid % 224); long t1 = gid / 224;
    int oy = (int)(t1 % 224);  long t2 = t1 / 224;
    int c  = (int)(t2 % 3);    int pi = (int)(t2 / 3);
    int b  = pi / TOPN;
    int sel = idx[pi];
    float y0 = anch[sel * 4 + 0], x0 = anch[sel * 4 + 1];
    float y1 = anch[sel * 4 + 2], x1 = anch[sel * 4 + 3];
    float ty = (float)oy / 223.f, tx = (float)ox / 223.f;
    float ys = y0 + ty * (y1 - y0 - 1.f);
    float xs = x0 + tx * (x1 - x0 - 1.f);
    float fy = floorf(ys), fx = floorf(xs);
    float wy = ys - fy, wx = xs - fx;
    int y0i = min(max((int)fy, 0), LPADW - 1), y1i = min(max((int)fy + 1, 0), LPADW - 1);
    int x0i = min(max((int)fx, 0), LPADW - 1), x1i = min(max((int)fx + 1, 0), LPADW - 1);
    auto g = [&](int yi, int xi) -> float {
        int yy = yi - PADW, xx = xi - PADW;
        if (yy < 0 || yy >= IMS || xx < 0 || xx >= IMS) return 0.f;
        return x[(((size_t)b * 3 + c) * IMS + yy) * IMS + xx];
    };
    float top = g(y0i, x0i) * (1.f - wx) + g(y0i, x1i) * wx;
    float bot = g(y1i, x0i) * (1.f - wx) + g(y1i, x1i) * wx;
    parts[gid] = top * (1.f - wy) + bot * wy;
}

// ============================================================
// AdaptiveMaxPool2d(1): [n, C, HW] -> [n, C]
// ============================================================
__global__ void pool_max_kernel(const float* __restrict__ in, float* __restrict__ out,
                                int n, int C, int HW)
{
    int i = blockIdx.x * blockDim.x + threadIdx.x;
    if (i >= n * C) return;
    const float* p = in + (size_t)i * HW;
    float m = p[0];
    for (int j = 1; j < HW; ++j) m = fmaxf(m, p[j]);
    out[i] = m;
}

// ============================================================
// Part ordering by descending L2 norm of f3p features
// ============================================================
__global__ __launch_bounds__(256)
void norm_order_kernel(const float* __restrict__ f3f, int* __restrict__ order)
{
    int b = blockIdx.x;
    __shared__ float red[256];
    __shared__ float nrm[TOPN];
    int tid = threadIdx.x;
    for (int t = 0; t < TOPN; ++t) {
        float s = 0.f;
        for (int c = tid; c < CH2; c += 256) {
            float v = f3f[(size_t)(b * TOPN + t) * CH2 + c];
            s += v * v;
        }
        red[tid] = s; __syncthreads();
        for (int st = 128; st > 0; st >>= 1) {
            if (tid < st) red[tid] += red[tid + st];
            __syncthreads();
        }
        if (tid == 0) nrm[t] = red[0];
        __syncthreads();
    }
    if (tid == 0) {
        bool used[TOPN] = {false, false, false, false};
        for (int rr = 0; rr < TOPN; ++rr) {
            int best = 0; float bv = -INFINITY;
            for (int t = 0; t < TOPN; ++t)
                if (!used[t] && nrm[t] > bv) { bv = nrm[t]; best = t; }
            used[best] = true;
            order[b * TOPN + rr] = best;
        }
    }
}

// gather part features by order into [B, TOPN*CH2] for the three reg-MLPs
__global__ void sel_kernel(const float* __restrict__ f1f, const float* __restrict__ f2f,
                           const float* __restrict__ f3f, const int* __restrict__ order,
                           float* __restrict__ s1, float* __restrict__ s2, float* __restrict__ s3)
{
    int i = blockIdx.x * blockDim.x + threadIdx.x;
    if (i >= BATCH * TOPN * CH2) return;
    int c = i % CH2;
    int t = (i / CH2) % TOPN;
    int b = i / (CH2 * TOPN);
    int src = (b * TOPN + order[b * TOPN + t]) * CH2 + c;
    s1[i] = f1f[src]; s2[i] = f2f[src]; s3[i] = f3f[src];
}

__global__ void concat3_kernel(const float* __restrict__ a, const float* __restrict__ b,
                               const float* __restrict__ c, float* __restrict__ dst, int n)
{
    int i = blockIdx.x * blockDim.x + threadIdx.x;
    if (i >= n * 3 * CH2) return;
    int col = i % (3 * CH2), row = i / (3 * CH2);
    float v;
    if (col < CH2)            v = a[row * CH2 + col];
    else if (col < 2 * CH2)   v = b[row * CH2 + col - CH2];
    else                      v = c[row * CH2 + col - 2 * CH2];
    dst[i] = v;
}

// ============================================================
// host-side launcher
// ============================================================
static inline void launch_gemm(hipStream_t st, const float* Wt, const float* X, float* Y,
                               const float* bias, const float* scale, const float* shift, float smul,
                               const float* iscale, const float* ishift, float imul, int act,
                               int M, int Cin, int H, int Wd, int KH, int KW, int stride, int pad,
                               int Bn, int OH, int OW, int obs, int ooff)
{
    int N = Bn * OH * OW;
    dim3 g((N + BN - 1) / BN, (M + BM - 1) / BM);
    gemm_conv_wmma<<<g, 256, 0, st>>>(Wt, X, Y, bias, scale, shift, smul,
                                      iscale, ishift, imul, act,
                                      M, Cin, H, Wd, KH, KW, stride, pad, Bn, OH, OW, obs, ooff);
}

extern "C" void kernel_launch(void* const* d_in, const int* in_sizes, int n_in,
                              void* d_out, int out_size, void* d_ws, size_t ws_size,
                              hipStream_t stream)
{
    (void)in_sizes; (void)n_in; (void)out_size; (void)ws_size;
    #define IN(i) ((const float*)d_in[i])
    const float* x = IN(0);
    float* out = (float*)d_out;
    const float rs = 1.0f / sqrtf(1.0f + EPSBN);   // eval BN: running mean 0 / var 1

    // ---- workspace layout (floats) ----
    float* ws = (float*)d_ws;
    size_t off = 0;
    auto alloc = [&](size_t nfl) { float* p = ws + off; off += nfl; return p; };
    float* anch   = alloc((size_t)NANCH * 4);
    float* rpn    = alloc((size_t)BATCH * NANCH);
    int*   idxb   = (int*)alloc(BATCH * TOPN);
    int*   order  = (int*)alloc(BATCH * TOPN);
    float* f1     = alloc((size_t)BATCH * CH4 * 56 * 56);
    float* f2     = alloc((size_t)BATCH * CH2 * 28 * 28);
    float* f3     = alloc((size_t)BATCH * CH  * 14 * 14);
    float* d1b    = alloc((size_t)BATCH * 128 * 14 * 14);
    float* d2b    = alloc((size_t)BATCH * 128 * 7 * 7);
    float* d3b    = alloc((size_t)BATCH * 128 * 4 * 4);
    float* parts  = alloc((size_t)NPART * 3 * 224 * 224);
    float* f1p    = alloc((size_t)NPART * CH4 * 28 * 28);
    float* f2p    = alloc((size_t)NPART * CH2 * 14 * 14);
    float* f3p    = alloc((size_t)NPART * CH  * 7 * 7);
    float* tmp1   = alloc((size_t)NPART * CH4 * 28 * 28);   // also fits [4,512,56,56]
    float* tmp2   = alloc((size_t)NPART * CH2 * 28 * 28);   // also fits [4,1024,56,56]
    float* f1pf   = alloc((size_t)NPART * CH2);
    float* f2pf   = alloc((size_t)NPART * CH2);
    float* f3pf   = alloc((size_t)NPART * CH2);
    float* catp   = alloc((size_t)NPART * 3 * CH2);
    float* catg   = alloc((size_t)BATCH * 3 * CH2);
    float* hid    = alloc((size_t)NPART * FEATD);
    float* sel1   = alloc((size_t)BATCH * TOPN * CH2);
    float* sel2   = alloc((size_t)BATCH * TOPN * CH2);
    float* sel3   = alloc((size_t)BATCH * TOPN * CH2);
    float* rhid   = alloc((size_t)BATCH * CH2);

    // ---- output slice offsets (floats) ----
    float* y1  = out + 0;       float* y2  = out + 800;
    float* y3  = out + 1600;    float* y4  = out + 2400;
    float* yp1 = out + 3200;    float* yp2 = out + 6400;
    float* yp3 = out + 9600;    float* yp4 = out + 12800;
    float* tnp = out + 16000;
    float* f1m = out + 16016;   float* f1g = out + 20112;
    float* f2m = out + 24208;   float* f2g = out + 28304;
    float* f3m = out + 32400;   float* f3g = out + 36496;

    // ---- anchors ----
    anchors_kernel<<<(NANCH + 255) / 256, 256, 0, stream>>>(anch);

    // ---- global backbone ----
    launch_gemm(stream, IN(1), x,  f1, 0, 0, 0, 1.f, 0, 0, 1.f, 1, CH4, 3,   448, 448, 8, 8, 8, 0, BATCH, 56, 56, CH4 * 3136, 0);
    launch_gemm(stream, IN(2), f1, f2, 0, 0, 0, 1.f, 0, 0, 1.f, 1, CH2, CH4, 56,  56,  3, 3, 2, 1, BATCH, 28, 28, CH2 * 784,  0);
    launch_gemm(stream, IN(3), f2, f3, 0, 0, 0, 1.f, 0, 0, 1.f, 1, CH,  CH2, 28,  28,  3, 3, 2, 1, BATCH, 14, 14, CH  * 196,  0);

    // ---- proposal net ----
    launch_gemm(stream, IN(4), f3,  d1b, IN(5), 0, 0, 1.f, 0, 0, 1.f, 1, 128, CH,  14, 14, 3, 3, 1, 1, BATCH, 14, 14, 128 * 196, 0);
    launch_gemm(stream, IN(6), d1b, d2b, IN(7), 0, 0, 1.f, 0, 0, 1.f, 1, 128, 128, 14, 14, 3, 3, 2, 1, BATCH, 7,  7,  128 * 49,  0);
    launch_gemm(stream, IN(8), d2b, d3b, IN(9), 0, 0, 1.f, 0, 0, 1.f, 1, 128, 128, 7,  7,  3, 3, 2, 1, BATCH, 4,  4,  128 * 16,  0);
    launch_gemm(stream, IN(10), d1b, rpn, IN(11), 0, 0, 1.f, 0, 0, 1.f, 0, 6, 128, 14, 14, 1, 1, 1, 0, BATCH, 14, 14, NANCH, 0);
    launch_gemm(stream, IN(12), d2b, rpn, IN(13), 0, 0, 1.f, 0, 0, 1.f, 0, 6, 128, 7,  7,  1, 1, 1, 0, BATCH, 7,  7,  NANCH, 1176);
    launch_gemm(stream, IN(14), d3b, rpn, IN(15), 0, 0, 1.f, 0, 0, 1.f, 0, 9, 128, 4,  4,  1, 1, 1, 0, BATCH, 4,  4,  NANCH, 1470);

    // ---- NMS + crops ----
    nms_kernel<<<BATCH, 256, 0, stream>>>(rpn, anch, idxb, tnp);
    long ctot = (long)NPART * 3 * 224 * 224;
    crop_kernel<<<(unsigned)((ctot + 255) / 256), 256, 0, stream>>>(x, anch, idxb, parts);

    // ---- part backbone ----
    launch_gemm(stream, IN(1), parts, f1p, 0, 0, 0, 1.f, 0, 0, 1.f, 1, CH4, 3,   224, 224, 8, 8, 8, 0, NPART, 28, 28, CH4 * 784, 0);
    launch_gemm(stream, IN(2), f1p,   f2p, 0, 0, 0, 1.f, 0, 0, 1.f, 1, CH2, CH4, 28,  28,  3, 3, 2, 1, NPART, 14, 14, CH2 * 196, 0);
    launch_gemm(stream, IN(3), f2p,   f3p, 0, 0, 0, 1.f, 0, 0, 1.f, 1, CH,  CH2, 14,  14,  3, 3, 2, 1, NPART, 7,  7,  CH  * 49,  0);

    // ---- conv blocks (shared weights): parts -> f?pf ; global -> f?g (in d_out) ----
    // parts cb1
    launch_gemm(stream, IN(16), f1p,  tmp1, 0, IN(17), IN(18), rs, 0, 0, 1.f, 1, FEATD, CH4, 28, 28, 1, 1, 1, 0, NPART, 28, 28, FEATD * 784, 0);
    launch_gemm(stream, IN(19), tmp1, tmp2, 0, IN(20), IN(21), rs, 0, 0, 1.f, 1, CH2, FEATD, 28, 28, 3, 3, 1, 1, NPART, 28, 28, CH2 * 784, 0);
    pool_max_kernel<<<(NPART * CH2 + 255) / 256, 256, 0, stream>>>(tmp2, f1pf, NPART, CH2, 784);
    // parts cb2
    launch_gemm(stream, IN(22), f2p,  tmp1, 0, IN(23), IN(24), rs, 0, 0, 1.f, 1, FEATD, CH2, 14, 14, 1, 1, 1, 0, NPART, 14, 14, FEATD * 196, 0);
    launch_gemm(stream, IN(25), tmp1, tmp2, 0, IN(26), IN(27), rs, 0, 0, 1.f, 1, CH2, FEATD, 14, 14, 3, 3, 1, 1, NPART, 14, 14, CH2 * 196, 0);
    pool_max_kernel<<<(NPART * CH2 + 255) / 256, 256, 0, stream>>>(tmp2, f2pf, NPART, CH2, 196);
    // parts cb3
    launch_gemm(stream, IN(28), f3p,  tmp1, 0, IN(29), IN(30), rs, 0, 0, 1.f, 1, FEATD, CH, 7, 7, 1, 1, 1, 0, NPART, 7, 7, FEATD * 49, 0);
    launch_gemm(stream, IN(31), tmp1, tmp2, 0, IN(32), IN(33), rs, 0, 0, 1.f, 1, CH2, FEATD, 7, 7, 3, 3, 1, 1, NPART, 7, 7, CH2 * 49, 0);
    pool_max_kernel<<<(NPART * CH2 + 255) / 256, 256, 0, stream>>>(tmp2, f3pf, NPART, CH2, 49);
    // global cb1
    launch_gemm(stream, IN(16), f1,   tmp1, 0, IN(17), IN(18), rs, 0, 0, 1.f, 1, FEATD, CH4, 56, 56, 1, 1, 1, 0, BATCH, 56, 56, FEATD * 3136, 0);
    launch_gemm(stream, IN(19), tmp1, tmp2, 0, IN(20), IN(21), rs, 0, 0, 1.f, 1, CH2, FEATD, 56, 56, 3, 3, 1, 1, BATCH, 56, 56, CH2 * 3136, 0);
    pool_max_kernel<<<(BATCH * CH2 + 255) / 256, 256, 0, stream>>>(tmp2, f1g, BATCH, CH2, 3136);
    // global cb2
    launch_gemm(stream, IN(22), f2,   tmp1, 0, IN(23), IN(24), rs, 0, 0, 1.f, 1, FEATD, CH2, 28, 28, 1, 1, 1, 0, BATCH, 28, 28, FEATD * 784, 0);
    launch_gemm(stream, IN(25), tmp1, tmp2, 0, IN(26), IN(27), rs, 0, 0, 1.f, 1, CH2, FEATD, 28, 28, 3, 3, 1, 1, BATCH, 28, 28, CH2 * 784, 0);
    pool_max_kernel<<<(BATCH * CH2 + 255) / 256, 256, 0, stream>>>(tmp2, f2g, BATCH, CH2, 784);
    // global cb3
    launch_gemm(stream, IN(28), f3,   tmp1, 0, IN(29), IN(30), rs, 0, 0, 1.f, 1, FEATD, CH, 14, 14, 1, 1, 1, 0, BATCH, 14, 14, FEATD * 196, 0);
    launch_gemm(stream, IN(31), tmp1, tmp2, 0, IN(32), IN(33), rs, 0, 0, 1.f, 1, CH2, FEATD, 14, 14, 3, 3, 1, 1, BATCH, 14, 14, CH2 * 196, 0);
    pool_max_kernel<<<(BATCH * CH2 + 255) / 256, 256, 0, stream>>>(tmp2, f3g, BATCH, CH2, 196);

    // ---- classifiers: base=param idx of g1 ; x -> bn1 -> fc1+bb1 -> elu(bn2) -> fc2+bb2 ----
    auto classifier = [&](int base, const float* xin, float* yout, int n, int din) {
        launch_gemm(stream, IN(base + 2), xin, hid, IN(base + 3), IN(base + 4), IN(base + 5), rs,
                    IN(base + 0), IN(base + 1), rs, 2,
                    FEATD, din, 1, 1, 1, 1, 1, 0, n, 1, 1, FEATD, 0);
        launch_gemm(stream, IN(base + 6), hid, yout, IN(base + 7), 0, 0, 1.f, 0, 0, 1.f, 0,
                    CLS, FEATD, 1, 1, 1, 1, 1, 0, n, 1, 1, CLS, 0);
    };
    concat3_kernel<<<(NPART * 3 * CH2 + 255) / 256, 256, 0, stream>>>(f1pf, f2pf, f3pf, catp, NPART);
    concat3_kernel<<<(BATCH * 3 * CH2 + 255) / 256, 256, 0, stream>>>(f1g, f2g, f3g, catg, BATCH);
    classifier(34, f1pf, yp1, NPART, CH2);
    classifier(42, f2pf, yp2, NPART, CH2);
    classifier(50, f3pf, yp3, NPART, CH2);
    classifier(58, catp, yp4, NPART, 3 * CH2);
    classifier(34, f1g,  y1,  BATCH, CH2);
    classifier(42, f2g,  y2,  BATCH, CH2);
    classifier(50, f3g,  y3,  BATCH, CH2);
    classifier(58, catg, y4,  BATCH, 3 * CH2);

    // ---- PartsResort + reg MLPs ----
    norm_order_kernel<<<BATCH, 256, 0, stream>>>(f3pf, order);
    sel_kernel<<<(BATCH * TOPN * CH2 + 255) / 256, 256, 0, stream>>>(f1pf, f2pf, f3pf, order, sel1, sel2, sel3);
    auto regmlp = [&](int base, const float* xin, float* yout) {
        launch_gemm(stream, IN(base + 0), xin, rhid, IN(base + 1), 0, 0, 1.f, 0, 0, 1.f, 2,
                    CH2, TOPN * CH2, 1, 1, 1, 1, 1, 0, BATCH, 1, 1, CH2, 0);
        launch_gemm(stream, IN(base + 2), rhid, yout, IN(base + 3), 0, 0, 1.f, 0, 0, 1.f, 0,
                    CH2, CH2, 1, 1, 1, 1, 1, 0, BATCH, 1, 1, CH2, 0);
    };
    regmlp(66, sel1, f1m);
    regmlp(70, sel2, f2m);
    regmlp(74, sel3, f3m);
    #undef IN
}
